// SLYOLO_32624571580790
// MI455X (gfx1250) — matrix operand order repaired and checked by reading, not compiled, not run on hardware
//
#include <hip/hip_runtime.h>
#include <hip/hip_bf16.h>
#include <math.h>

// ---------------- problem constants ----------------
#define HW     65536          // H*W
#define WIMG   256
#define KDIM   256            // channels
#define NDIM   256            // packed outputs (240 cls + 3 obj + 12 reg + 1 pad)
#define MTILE  32             // pixels per block
#define KP     264            // padded LDS row (halves) to break bank conflicts
#define NCLS   80
#define NANCH  3
#define MROWS  (HW * NANCH)   // 196608 anchors
#define TOPKN  1000
#define NBINS  8192
#define NCAND  4096
#define CONF_T 0.05f
#define NMS_T  0.6f
#define IMGSZ  2048.0f        // max(H,W)*stride

typedef __attribute__((ext_vector_type(16))) _Float16 v16h;
typedef __attribute__((ext_vector_type(8)))  _Float16 v8h;
typedef __attribute__((ext_vector_type(8)))  float    v8f;

__device__ __forceinline__ float sigm(float x) { return 1.0f / (1.0f + __expf(-x)); }

// A fragment (16x32 f16): lanes 0-15 hold K {k0..k0+7, k0+16..k0+23} of row laneM;
// lanes 16-31 hold the +8 K-offset. Rows come from padded LDS tile (stride KP).
__device__ __forceinline__ v16h loadA(const _Float16* base, int row, int k0, int laneHi) {
    const _Float16* pa = base + row * KP + k0 + 8 * laneHi;
    v8h lo = *(const v8h*)pa;
    v8h hi = *(const v8h*)(pa + 16);
    return __builtin_shufflevector(lo, hi, 0,1,2,3,4,5,6,7,8,9,10,11,12,13,14,15);
}

// B fragment (32x16 f16): lane owns one N column, K {k0+8h..+7, k0+16+8h..+7}.
__device__ __forceinline__ v16h loadB(const _Float16* wall, int nCol, int k0, int laneHi) {
    const _Float16* pb = wall + (size_t)nCol * KDIM + k0 + 8 * laneHi;
    v8h lo = *(const v8h*)pb;
    v8h hi = *(const v8h*)(pb + 16);
    return __builtin_shufflevector(lo, hi, 0,1,2,3,4,5,6,7,8,9,10,11,12,13,14,15);
}

// ---------------- workspace layout ----------------
__host__ __device__ constexpr size_t alignUp(size_t x) { return (x + 255) & ~(size_t)255; }
constexpr size_t WS_WALL = 0;                                        // f16 [256][256]
constexpr size_t WS_BALL = alignUp(WS_WALL + (size_t)NDIM * KDIM * 2);
constexpr size_t WS_CONV = alignUp(WS_BALL + (size_t)NDIM * 4);      // f32 [65536][256]
constexpr size_t WS_HIST = alignUp(WS_CONV + (size_t)HW * NDIM * 4); // u32 [NBINS]
constexpr size_t WS_CNT  = WS_HIST + (size_t)NBINS * 4;              // u32 counter (contiguous w/ hist for 1 memset)
constexpr size_t WS_THB  = alignUp(WS_CNT + 4);                      // i32 threshold bin
constexpr size_t WS_CSC  = alignUp(WS_THB + 4);                      // f32 [NCAND]
constexpr size_t WS_CIX  = alignUp(WS_CSC + (size_t)NCAND * 4);      // i32 [NCAND]
constexpr size_t WS_SIDX = alignUp(WS_CIX + (size_t)NCAND * 4);      // i32 [TOPK] sorted indices
constexpr size_t WS_BOX  = alignUp(WS_SIDX + (size_t)TOPKN * 4);     // f32 [TOPK][4] raw boxes
constexpr size_t WS_LAB  = alignUp(WS_BOX + (size_t)TOPKN * 16);     // i32 [TOPK]

// ============ 1) pack weights (fp32 -> f16, fused layout) ============
// column order: 0..239 = cls (a*80+c), 240..242 = obj, 243..254 = reg, 255 = pad
__global__ void pack_w(const float* __restrict__ wobj, const float* __restrict__ bobj,
                       const float* __restrict__ wcls, const float* __restrict__ bcls,
                       const float* __restrict__ wreg, const float* __restrict__ breg,
                       _Float16* __restrict__ wall, float* __restrict__ ball) {
    int n = blockIdx.x, k = threadIdx.x;
    float wv = 0.0f, bv = 0.0f;
    if (n < 240)      { wv = wcls[(size_t)n * KDIM + k];        bv = bcls[n]; }
    else if (n < 243) { wv = wobj[(size_t)(n - 240) * KDIM + k]; bv = bobj[n - 240]; }
    else if (n < 255) { wv = wreg[(size_t)(n - 243) * KDIM + k]; bv = breg[n - 243]; }
    wall[(size_t)n * KDIM + k] = (_Float16)wv;
    if (k == 0) ball[n] = bv;
}

// ============ 2) fused 1x1-conv GEMM via WMMA ============
// out[p][n] = sum_k feat[k][p] * W[n][k] + b[n];  feat chosen per n-tile.
// Block tile: M=32 pixels x N=256 outputs x K=256. 8 waves; each wave owns
// two 16-col N tiles and both 16-row M halves -> 4 accumulators, 32 WMMAs.
__global__ __launch_bounds__(256) void gemm_wmma(
        const float* __restrict__ cls_feats, const float* __restrict__ reg_feats,
        const _Float16* __restrict__ wall, const float* __restrict__ ball,
        float* __restrict__ out) {
    __shared__ _Float16 Acls[MTILE][KP];
    __shared__ _Float16 Areg[MTILE][KP];

    const size_t p0 = (size_t)blockIdx.x * MTILE;
    const int m  = threadIdx.x & 31;
    const int c0 = (threadIdx.x >> 5) * 32;

    // stage A tiles: coalesced 128B rows per channel, fp32 -> f16 into LDS
    #pragma unroll 8
    for (int j = 0; j < 32; ++j) {
        size_t g = (size_t)(c0 + j) * HW + p0 + m;
        Acls[m][c0 + j] = (_Float16)cls_feats[g];
        Areg[m][c0 + j] = (_Float16)reg_feats[g];
    }
    // speculative prefetch of the next block's pixels (global_prefetch_b8)
    {
        size_t gn = (size_t)c0 * HW + p0 + MTILE + m;
        __builtin_prefetch(&cls_feats[gn], 0, 0);
        __builtin_prefetch(&reg_feats[gn], 0, 0);
    }
    __syncthreads();

    const int lane   = threadIdx.x & 31;
    const int laneM  = lane & 15;
    const int laneHi = lane >> 4;
    const int wv     = threadIdx.x >> 5;       // wave id 0..7
    const int t1 = wv * 2 + 1;                 // second N tile index
    const int Nb0 = wv * 32, Nb1 = wv * 32 + 16;
    // tile 15 (obj+reg columns) uses reg_feats; everything else cls_feats
    const _Float16* A0 = &Acls[0][0];
    const _Float16* A1 = (t1 == 15) ? &Areg[0][0] : &Acls[0][0];

    v8f acc00 = {}, acc01 = {}, acc10 = {}, acc11 = {};
    #pragma unroll
    for (int k0 = 0; k0 < KDIM; k0 += 32) {
        v16h a0m0 = loadA(A0, laneM,      k0, laneHi);
        v16h a0m1 = loadA(A0, laneM + 16, k0, laneHi);
        v16h a1m0 = loadA(A1, laneM,      k0, laneHi);
        v16h a1m1 = loadA(A1, laneM + 16, k0, laneHi);
        v16h b0   = loadB(wall, Nb0 + laneM, k0, laneHi);
        v16h b1   = loadB(wall, Nb1 + laneM, k0, laneHi);

        acc00 = __builtin_amdgcn_wmma_f32_16x16x32_f16(false, a0m0, false, b0, (short)0, acc00, false, false);
        acc10 = __builtin_amdgcn_wmma_f32_16x16x32_f16(false, a0m1, false, b0, (short)0, acc10, false, false);
        acc01 = __builtin_amdgcn_wmma_f32_16x16x32_f16(false, a1m0, false, b1, (short)0, acc01, false, false);
        acc11 = __builtin_amdgcn_wmma_f32_16x16x32_f16(false, a1m1, false, b1, (short)0, acc11, false, false);
    }

    // epilogue: C/D layout -> lane is one N column; VGPR r = rows (r, r+8)
    const int mBase = laneHi * 8;
    {
        int nc = Nb0 + laneM;
        float bias = ball[nc];
        #pragma unroll
        for (int r = 0; r < 8; ++r) {
            out[(p0 + mBase + r) * NDIM + nc]      = acc00[r] + bias;
            out[(p0 + 16 + mBase + r) * NDIM + nc] = acc10[r] + bias;
        }
    }
    {
        int nc = Nb1 + laneM;
        float bias = ball[nc];
        #pragma unroll
        for (int r = 0; r < 8; ++r) {
            out[(p0 + mBase + r) * NDIM + nc]      = acc01[r] + bias;
            out[(p0 + 16 + mBase + r) * NDIM + nc] = acc11[r] + bias;
        }
    }
}

// ============ 3) score histogram (LDS-local, merged globally) ============
__global__ __launch_bounds__(256) void score_hist(const float* __restrict__ conv,
                                                  unsigned* __restrict__ hist) {
    __shared__ unsigned sh[NBINS];
    for (int i = threadIdx.x; i < NBINS; i += 256) sh[i] = 0u;
    __syncthreads();

    int row = blockIdx.x * 256 + threadIdx.x;   // anchor row = p*3 + a
    if (row < MROWS) {
        int p = row / NANCH, a = row % NANCH;
        const float* rowp = conv + (size_t)p * NDIM;
        float so = sigm(rowp[240 + a]);
        for (int c = 0; c < NCLS; ++c) {
            float s = __fsqrt_rn(so * sigm(rowp[a * NCLS + c]));
            int bin = (int)(s * NBINS);
            bin = bin < 0 ? 0 : (bin >= NBINS ? NBINS - 1 : bin);
            atomicAdd(&sh[bin], 1u);
        }
    }
    __syncthreads();
    for (int i = threadIdx.x; i < NBINS; i += 256)
        if (sh[i]) atomicAdd(&hist[i], sh[i]);
}

// ============ 4) pick threshold bin ============
__global__ void pick_thresh(const unsigned* __restrict__ hist, int* __restrict__ thb) {
    unsigned suffix = 0;
    int b = NBINS - 1;
    for (; b >= 0; --b) {
        suffix += hist[b];
        if (suffix >= TOPKN) break;
    }
    if (b < 0) b = 0;
    // protect candidate cap: raise threshold (incrementally) if tail overflows NCAND
    while (b < NBINS - 1 && suffix > (unsigned)NCAND) {
        suffix -= hist[b];
        ++b;
    }
    *thb = b;
}

// ============ 5) collect candidates >= threshold ============
__global__ __launch_bounds__(256) void collect(const float* __restrict__ conv,
                                               const int* __restrict__ thb,
                                               unsigned* __restrict__ cnt,
                                               float* __restrict__ csc,
                                               int* __restrict__ cix) {
    int row = blockIdx.x * 256 + threadIdx.x;
    if (row >= MROWS) return;
    int p = row / NANCH, a = row % NANCH;
    const float* rowp = conv + (size_t)p * NDIM;
    float so = sigm(rowp[240 + a]);
    int tb = *thb;
    for (int c = 0; c < NCLS; ++c) {
        float s = __fsqrt_rn(so * sigm(rowp[a * NCLS + c]));
        int bin = (int)(s * NBINS);
        bin = bin < 0 ? 0 : (bin >= NBINS ? NBINS - 1 : bin);
        if (bin >= tb) {
            unsigned slot = atomicAdd(cnt, 1u);
            if (slot < NCAND) { csc[slot] = s; cix[slot] = row * NCLS + c; }
        }
    }
}

// ============ 6) one-block bitonic sort + select top-1000 ============
__global__ __launch_bounds__(1024) void sort_select(const unsigned* __restrict__ cnt,
                                                    const float* __restrict__ csc,
                                                    const int* __restrict__ cix,
                                                    float* __restrict__ outScores,
                                                    int* __restrict__ sidx) {
    __shared__ float sv[NCAND];
    __shared__ int   si[NCAND];
    int n = (int)*cnt; if (n > NCAND) n = NCAND;
    for (int t = threadIdx.x; t < NCAND; t += 1024) {
        if (t < n) { sv[t] = csc[t]; si[t] = cix[t]; }
        else       { sv[t] = -1.0f;  si[t] = 0x7fffffff; }
    }
    __syncthreads();
    // ascending by (-score, idx): position 0 = highest score, ties -> smaller idx
    for (int k = 2; k <= NCAND; k <<= 1) {
        for (int j = k >> 1; j > 0; j >>= 1) {
            for (int t = threadIdx.x; t < NCAND; t += 1024) {
                int x = t ^ j;
                if (x > t) {
                    bool asc = ((t & k) == 0);
                    float s1 = sv[t], s2 = sv[x];
                    int   i1 = si[t], i2 = si[x];
                    bool beforeX = (s2 > s1) || (s2 == s1 && i2 < i1); // x should come first
                    if (asc == beforeX) { sv[t] = s2; si[t] = i2; sv[x] = s1; si[x] = i1; }
                }
            }
            __syncthreads();
        }
    }
    for (int t = threadIdx.x; t < TOPKN; t += 1024) {
        outScores[t] = sv[t];
        sidx[t] = si[t];
    }
}

// ============ 7) decode boxes ============
__global__ __launch_bounds__(1024) void decode(const float* __restrict__ conv,
                                               const int* __restrict__ sidx,
                                               float* __restrict__ boxraw,
                                               int* __restrict__ lab) {
    const float AW[3] = {10.0f, 16.0f, 33.0f};
    const float AH[3] = {13.0f, 30.0f, 23.0f};
    int k = threadIdx.x;
    if (k >= TOPKN) return;
    int idx = sidx[k];
    if (idx < 0 || idx >= MROWS * NCLS) idx = 0;   // sentinel -> keep=false later
    int row = idx / NCLS, cl = idx % NCLS;
    int p = row / NANCH, a = row % NANCH;
    const float* rowp = conv + (size_t)p * NDIM + 243 + a * 4;
    float r0 = rowp[0], r1 = rowp[1], r2 = rowp[2], r3 = rowp[3];
    float px = (float)(p % WIMG) + 0.5f;
    float py = (float)(p / WIMG) + 0.5f;
    float cx = (px + sigm(r0)) * 8.0f;
    float cy = (py + sigm(r1)) * 8.0f;
    float w  = __expf(r2) * AW[a];
    float h  = __expf(r3) * AH[a];
    boxraw[k * 4 + 0] = cx - 0.5f * w;
    boxraw[k * 4 + 1] = cy - 0.5f * h;
    boxraw[k * 4 + 2] = cx + 0.5f * w;
    boxraw[k * 4 + 3] = cy + 0.5f * h;
    lab[k] = cl;
}

// ============ 8) greedy NMS + final outputs ============
__global__ __launch_bounds__(1024) void nms_finalize(const float* __restrict__ boxraw,
                                                     const int* __restrict__ lab,
                                                     const float* __restrict__ scores,
                                                     float* __restrict__ dout) {
    __shared__ float x1[TOPKN], y1[TOPKN], x2[TOPKN], y2[TOPKN], ar[TOPKN];
    __shared__ int   sl[TOPKN], sk[TOPKN];
    for (int k = threadIdx.x; k < TOPKN; k += 1024) {
        float a = boxraw[k * 4 + 0], b = boxraw[k * 4 + 1];
        float c = boxraw[k * 4 + 2], d = boxraw[k * 4 + 3];
        x1[k] = a; y1[k] = b; x2[k] = c; y2[k] = d;
        ar[k] = (c - a) * (d - b);
        sl[k] = lab[k];
        sk[k] = (scores[k] > CONF_T) ? 1 : 0;
    }
    __syncthreads();
    for (int i = 0; i < TOPKN; ++i) {
        if (sk[i]) {
            for (int j = threadIdx.x; j < TOPKN; j += 1024) {
                if (j > i && sk[j] && sl[j] == sl[i]) {
                    float iw = fminf(x2[i], x2[j]) - fmaxf(x1[i], x1[j]);
                    float ih = fminf(y2[i], y2[j]) - fmaxf(y1[i], y1[j]);
                    iw = fmaxf(1e-10f, iw); ih = fmaxf(1e-10f, ih);
                    float inter = iw * ih;
                    float iou = inter / (ar[i] + ar[j] - inter + 1e-14f);
                    if (iou > NMS_T) sk[j] = 0;
                }
            }
        }
        __syncthreads();
    }
    // outputs: bboxes[1000*4] | scores[1000] (already written) | labels[1000] | keep[1000]
    for (int k = threadIdx.x; k < TOPKN; k += 1024) {
        #pragma unroll
        for (int c = 0; c < 4; ++c) {
            float v = boxraw[k * 4 + c] * (1.0f / IMGSZ);
            v = fminf(fmaxf(v, 0.0f), 1.0f);
            dout[k * 4 + c] = v;
        }
        dout[5 * TOPKN + k] = (float)sl[k];
        dout[6 * TOPKN + k] = sk[k] ? 1.0f : 0.0f;
    }
}

// ---------------- host launch ----------------
extern "C" void kernel_launch(void* const* d_in, const int* in_sizes, int n_in,
                              void* d_out, int out_size, void* d_ws, size_t ws_size,
                              hipStream_t stream) {
    const float* cls_feats = (const float*)d_in[0];
    const float* reg_feats = (const float*)d_in[1];
    const float* w_obj = (const float*)d_in[2];
    const float* b_obj = (const float*)d_in[3];
    const float* w_cls = (const float*)d_in[4];
    const float* b_cls = (const float*)d_in[5];
    const float* w_reg = (const float*)d_in[6];
    const float* b_reg = (const float*)d_in[7];
    float* dout = (float*)d_out;
    char* ws = (char*)d_ws;

    _Float16* wall = (_Float16*)(ws + WS_WALL);
    float*    ball = (float*)(ws + WS_BALL);
    float*    conv = (float*)(ws + WS_CONV);
    unsigned* hist = (unsigned*)(ws + WS_HIST);
    unsigned* cnt  = (unsigned*)(ws + WS_CNT);
    int*      thb  = (int*)(ws + WS_THB);
    float*    csc  = (float*)(ws + WS_CSC);
    int*      cix  = (int*)(ws + WS_CIX);
    int*      sidx = (int*)(ws + WS_SIDX);
    float*    boxr = (float*)(ws + WS_BOX);
    int*      lab  = (int*)(ws + WS_LAB);

    // zero histogram + counter (graph-capture-safe async memset)
    hipMemsetAsync(ws + WS_HIST, 0, (size_t)NBINS * 4 + 256, stream);

    pack_w<<<NDIM, KDIM, 0, stream>>>(w_obj, b_obj, w_cls, b_cls, w_reg, b_reg, wall, ball);
    gemm_wmma<<<HW / MTILE, 256, 0, stream>>>(cls_feats, reg_feats, wall, ball, conv);
    score_hist<<<MROWS / 256, 256, 0, stream>>>(conv, hist);
    pick_thresh<<<1, 1, 0, stream>>>(hist, thb);
    collect<<<MROWS / 256, 256, 0, stream>>>(conv, thb, cnt, csc, cix);
    sort_select<<<1, 1024, 0, stream>>>(cnt, csc, cix, dout + 4 * TOPKN, sidx);
    decode<<<1, 1024, 0, stream>>>(conv, sidx, boxr, lab);
    nms_finalize<<<1, 1024, 0, stream>>>(boxr, lab, dout + 4 * TOPKN, dout);
}
